// SpatialBlock_89781996356062
// MI455X (gfx1250) — matrix operand order repaired and checked by reading, not compiled, not run on hardware
//
#include <hip/hip_runtime.h>
#include <hip/hip_bf16.h>

// ---------------- problem constants (from reference) ----------------
#define B_ 16
#define C_ 64
#define T_ 128
#define D_ 256
#define H_ 8
#define KD_ 32

typedef __attribute__((ext_vector_type(16))) _Float16 v16h;
typedef __attribute__((ext_vector_type(8)))  _Float16 v8h;
typedef __attribute__((ext_vector_type(8)))  float    v8f;

// Build a v16h WMMA operand from two contiguous 8-half loads.
__device__ __forceinline__ v16h cat8(v8h lo, v8h hi) {
    v16h r;
#pragma unroll
    for (int e = 0; e < 8; ++e) { r[e] = lo[e]; r[e + 8] = hi[e]; }
    return r;
}

// A operand (16x32 f16) from row-major [M][K] with row stride S.
// gfx1250 layout: lane<16 -> row m0+lane, K = k0+{0..7,16..23};
//                 lane>=16 -> row m0+lane-16, K = k0+{8..15,24..31}.
__device__ __forceinline__ v16h load_a16x32(const _Float16* base, int S,
                                            int m0, int k0, int lane) {
    const int hi  = lane >> 4;
    const int row = m0 + (lane & 15);
    const _Float16* p = base + row * S;
    v8h lo8 = *(const v8h*)(p + k0 + (hi ? 8  : 0));
    v8h hi8 = *(const v8h*)(p + k0 + (hi ? 24 : 16));
    return cat8(lo8, hi8);
}

// B operand (32x16 f16) from K-contiguous layout: element (k,n) at base[n*KS + k].
// gfx1250 layout: lanes 0-15 hold K=k0+{0..7,16..23} (col n0+lane), lanes 16-31 K=k0+{8..15,24..31}.
__device__ __forceinline__ v16h load_b32x16(const _Float16* base, int KS,
                                            int k0, int n0, int lane) {
    const int hi  = lane >> 4;
    const int col = n0 + (lane & 15);
    const _Float16* p = base + col * KS + k0 + (hi ? 8 : 0);
    v8h lo8 = *(const v8h*)(p);
    v8h hi8 = *(const v8h*)(p + 16);
    return cat8(lo8, hi8);
}

// ---------------- kernel A: h = mean over T ----------------
__global__ __launch_bounds__(256) void mean_kernel(const float* __restrict__ x,
                                                   float* __restrict__ hmean) {
    const int bc = blockIdx.x;          // 0..B*C-1
    const int d  = threadIdx.x;         // 0..255
    const float* p = x + (size_t)bc * T_ * D_ + d;
    float s = 0.f;
#pragma unroll 4
    for (int t = 0; t < T_; ++t) s += p[(size_t)t * D_];
    hmean[(size_t)bc * D_ + d] = s * (1.0f / (float)T_);
}

// ---------------- kernel B: dynamic adjacency, delta_a, top-k mask --------
__global__ __launch_bounds__(256) void adj_kernel(
    const float* __restrict__ hmean, const float* __restrict__ sadj,
    const float* __restrict__ wq, const float* __restrict__ wk,
    float* __restrict__ delta_out, unsigned long long* __restrict__ mask_out) {
    __shared__ float qs[C_ * KD_];
    __shared__ float ks[C_ * KD_];
    __shared__ float ab[C_ * C_];   // |static + delta| (destroyed by top-k)

    const int b = blockIdx.x, tid = threadIdx.x;
    const float* hb = hmean + (size_t)b * C_ * D_;

    // q = h @ wq^T, k = h @ wk^T   (64x32 each; 4096 dot products / 256 thr)
    for (int u = 0; u < 16; ++u) {
        int g = tid + 256 * u;                // 0..4095
        int which = g >> 11;                  // 0 -> q, 1 -> k
        int rem = g & 2047;
        int c = rem >> 5, kk = rem & 31;
        const float* w = (which ? wk : wq) + kk * D_;
        const float* hr = hb + c * D_;
        float s = 0.f;
        for (int d = 0; d < D_; ++d) s += hr[d] * w[d];
        if (which) ks[c * KD_ + kk] = s; else qs[c * KD_ + kk] = s;
    }
    __syncthreads();

    const float scale = 0.17677669529663687f;   // 1/sqrt(32)
    for (int u = 0; u < 16; ++u) {
        int g = tid + 256 * u;                // (i,j)
        int i = g >> 6, j = g & 63;
        float s = 0.f;
        for (int kk = 0; kk < KD_; ++kk) s += qs[i * KD_ + kk] * ks[j * KD_ + kk];
        float dlt = tanhf(s * scale);
        delta_out[((size_t)b * C_ + i) * C_ + j] = dlt;
        ab[i * C_ + j] = fabsf(sadj[i * C_ + j] + dlt);
    }
    __syncthreads();

    // top-8 per source row i -> bit j set; plus self loop
    if (tid < C_) {
        const int i = tid;
        unsigned long long m = 1ull << i;
        for (int pick = 0; pick < 8; ++pick) {
            float best = -1.f; int bj = 0;
            for (int j = 0; j < C_; ++j) {
                float v = ab[i * C_ + j];
                if (v > best) { best = v; bj = j; }
            }
            m |= 1ull << bj;
            ab[i * C_ + bj] = -1.f;
        }
        mask_out[b * C_ + i] = m;
    }
}

// ------------- kernel C: fold attention vectors through gat_lin -----------
// v_src[h,d] = sum_f att_src[h,f] * gat_lin[h*D+f, d]   (so a_s = x . v_src)
__global__ __launch_bounds__(256) void vvec_kernel(
    const float* __restrict__ glin, const float* __restrict__ asrc,
    const float* __restrict__ adst, float* __restrict__ vsrc,
    float* __restrict__ vdst) {
    int g = blockIdx.x * 256 + threadIdx.x;   // 0..4095
    int which = g >> 11;
    int rem = g & 2047;
    int h = rem >> 8, d = rem & 255;
    const float* att = (which ? adst : asrc) + h * D_;
    float s = 0.f;
    for (int f = 0; f < D_; ++f) s += att[f] * glin[((size_t)(h * D_ + f)) * D_ + d];
    (which ? vdst : vsrc)[h * D_ + d] = s;
}

// ------------- kernel D: f16 copy of gat_lin, pre-scaled by 1/H -----------
__global__ __launch_bounds__(256) void wcvt_kernel(const float* __restrict__ glin,
                                                   _Float16* __restrict__ wf) {
    int g = blockIdx.x * 256 + threadIdx.x;
    wf[g] = (_Float16)(glin[g] * (1.0f / (float)H_));
}

// ---------------- main fused GAT kernel: one workgroup per (b,t) ----------
__global__ __launch_bounds__(256) void gat_main_kernel(
    const float* __restrict__ x, const float* __restrict__ gat_bias,
    const float* __restrict__ vsrc_g, const float* __restrict__ vdst_g,
    const unsigned long long* __restrict__ mask_g,
    const _Float16* __restrict__ wf,          // [H*D][D] f16, scaled by 1/H
    float* __restrict__ out) {
    __shared__ _Float16 xT[D_ * C_];          // x^T: [d][c]  (K-contig B operand)
    __shared__ _Float16 zb[C_ * 128];         // z chunk: [c][d_local] (A operand)
    __shared__ _Float16 alphaT[C_ * C_];      // alpha^T: [j][i] (A operand)
    __shared__ float as_[C_ * H_];
    __shared__ float ad_[C_ * H_];
    __shared__ unsigned long long mk[C_];

    const int tid  = threadIdx.x;
    const int lane = tid & 31;
    const int wave = tid >> 5;
    const int b = blockIdx.x >> 7;
    const int t = blockIdx.x & 127;

    // ---- stage 0: load x[b,:,t,:] -> xT (f16), coalesced global reads ----
#pragma unroll 4
    for (int c = 0; c < C_; ++c) {
        float v = x[(((size_t)(b * C_ + c)) * T_ + t) * D_ + tid];
        xT[tid * C_ + c] = (_Float16)v;       // tid == d
    }
    if (tid < C_) mk[tid] = mask_g[b * C_ + tid];
    __syncthreads();

    // ---- stage 1: attention logits a_s/a_d = x . v_{src,dst} ----
    for (int u = 0; u < 4; ++u) {
        int g = tid + 256 * u;                // 0..1023
        int c = g >> 4; int rem = g & 15; int h = rem >> 1; int which = rem & 1;
        const float* vv = (which ? vdst_g : vsrc_g) + h * D_;
        float s = 0.f;
        for (int d = 0; d < D_; ++d) s += (float)xT[d * C_ + c] * vv[d];
        if (which) ad_[c * H_ + h] = s; else as_[c * H_ + h] = s;
    }
    __syncthreads();

    // persistent out accumulators: 64 tiles of 16x16 over [C=64][D=256]; 8/wave
    v8f acc[8];
    {
        v8f zz = {};
#pragma unroll
        for (int i = 0; i < 8; ++i) acc[i] = zz;
    }

    for (int h = 0; h < H_; ++h) {
        const _Float16* wh = wf + (size_t)h * D_ * D_;
        // warm L2/L0 with this head's weights while we do softmax
#pragma unroll
        for (int u = 0; u < 4; ++u)
            __builtin_prefetch((const void*)(wh + tid * D_ + u * 64), 0, 1);

        // ---- masked softmax over sources i, per target j (threads 0..63) ----
        if (tid < C_) {
            const int j = tid;
            const float adj = ad_[j * H_ + h];
            float m = -3.4e38f;
            for (int i = 0; i < C_; ++i) {
                if ((mk[i] >> j) & 1ull) {
                    float e = as_[i * H_ + h] + adj;
                    e = e > 0.f ? e : 0.2f * e;   // leaky_relu
                    m = e > m ? e : m;
                }
            }
            float ssum = 0.f;
            for (int i = 0; i < C_; ++i) {
                if ((mk[i] >> j) & 1ull) {
                    float e = as_[i * H_ + h] + adj;
                    e = e > 0.f ? e : 0.2f * e;
                    ssum += __expf(e - m);
                }
            }
            const float inv = 1.f / ssum;       // self-loop guarantees ssum>0
            for (int i = 0; i < C_; ++i) {
                float a = 0.f;
                if ((mk[i] >> j) & 1ull) {
                    float e = as_[i * H_ + h] + adj;
                    e = e > 0.f ? e : 0.2f * e;
                    a = __expf(e - m) * inv;
                }
                alphaT[j * C_ + i] = (_Float16)a;
            }
        }
        __syncthreads();

        // process d-dimension in two 128-wide chunks to keep LDS < 64KB
        for (int chunk = 0; chunk < 2; ++chunk) {
            // ---- GEMM1: z[:,chunk] = alpha^T (64x64) @ x (64x128) ----
            // 32 output tiles (4m x 8n), 4 per wave
#pragma unroll
            for (int tt = 0; tt < 4; ++tt) {
                int idx = wave * 4 + tt;
                int m0  = (idx >> 3) << 4;
                int n0l = (idx & 7) << 4;
                int n0g = chunk * 128 + n0l;
                v8f c = {};
#pragma unroll
                for (int kk = 0; kk < 2; ++kk) {
                    v16h a  = load_a16x32(alphaT, C_, m0, kk * 32, lane);
                    v16h bm = load_b32x16(xT, C_, kk * 32, n0g, lane);
                    c = __builtin_amdgcn_wmma_f32_16x16x32_f16(
                            false, a, false, bm, (short)0, c, false, false);
                }
                const int col  = n0l + (lane & 15);
                const int roff = (lane < 16) ? 0 : 8;
#pragma unroll
                for (int r = 0; r < 8; ++r)
                    zb[(m0 + roff + r) * 128 + col] = (_Float16)c[r];
            }
            __syncthreads();

            // ---- GEMM2: out += z_chunk (64x128) @ W_h^T-chunk (128x256) ----
            // 64 output tiles (4m x 16n), 8 per wave, K = 128 per chunk
#pragma unroll
            for (int tt = 0; tt < 8; ++tt) {
                int idx = wave * 8 + tt;
                int m0 = (idx >> 4) << 4;
                int n0 = (idx & 15) << 4;
                v8f c = acc[tt];
#pragma unroll
                for (int k0 = 0; k0 < 128; k0 += 32) {
                    v16h a  = load_a16x32(zb, 128, m0, k0, lane);
                    v16h bm = load_b32x16(wh, D_, chunk * 128 + k0, n0, lane);
                    c = __builtin_amdgcn_wmma_f32_16x16x32_f16(
                            false, a, false, bm, (short)0, c, false, false);
                }
                acc[tt] = c;
            }
            __syncthreads();
        }
    }

    // ---- store: out[b, c, t, f] = acc + bias  (heads already averaged) ----
#pragma unroll
    for (int tt = 0; tt < 8; ++tt) {
        int idx = wave * 8 + tt;
        int m0 = (idx >> 4) << 4;
        int n0 = (idx & 15) << 4;
        const int col  = n0 + (lane & 15);
        const float bias = gat_bias[col];
        const int roff = (lane < 16) ? 0 : 8;
#pragma unroll
        for (int r = 0; r < 8; ++r) {
            int crow = m0 + roff + r;
            out[(((size_t)(b * C_ + crow)) * T_ + t) * D_ + col] = acc[tt][r] + bias;
        }
    }
}

// ---------------- host launcher ----------------
extern "C" void kernel_launch(void* const* d_in, const int* in_sizes, int n_in,
                              void* d_out, int out_size, void* d_ws, size_t ws_size,
                              hipStream_t stream) {
    (void)in_sizes; (void)n_in; (void)out_size; (void)ws_size;
    const float* x    = (const float*)d_in[0];
    const float* sadj = (const float*)d_in[1];
    const float* wq   = (const float*)d_in[2];
    const float* wk   = (const float*)d_in[3];
    const float* glin = (const float*)d_in[4];
    const float* asrc = (const float*)d_in[5];
    const float* adst = (const float*)d_in[6];
    const float* gb   = (const float*)d_in[7];

    // workspace layout (≈2.02 MB)
    char* ws = (char*)d_ws;
    float*              hmean = (float*)(ws);                       // 1 MB
    float*              vsrc  = (float*)(ws + 1048576);             // 8 KB
    float*              vdst  = (float*)(ws + 1048576 + 8192);      // 8 KB
    unsigned long long* mbits = (unsigned long long*)(ws + 1048576 + 16384); // 8 KB
    _Float16*           wf    = (_Float16*)(ws + 1048576 + 24576);  // 1 MB

    float* out_x     = (float*)d_out;                         // [B,C,T,D]
    float* out_delta = out_x + (size_t)B_ * C_ * T_ * D_;     // [B,C,C]

    mean_kernel<<<B_ * C_, 256, 0, stream>>>(x, hmean);
    adj_kernel<<<B_, 256, 0, stream>>>(hmean, sadj, wq, wk, out_delta, mbits);
    vvec_kernel<<<16, 256, 0, stream>>>(glin, asrc, adst, vsrc, vdst);
    wcvt_kernel<<<(H_ * D_ * D_) / 256, 256, 0, stream>>>(glin, wf);
    gat_main_kernel<<<B_ * T_, 256, 0, stream>>>(x, gb, vsrc, vdst, mbits, wf, out_x);
}
